// SA_Layer_17154099380606
// MI455X (gfx1250) — matrix operand order repaired and checked by reading, not compiled, not run on hardware
//
#include <hip/hip_runtime.h>

// ---------------------------------------------------------------------------
// SA_Layer (PCT self-attention), B=4, C=256, N=4096, d_k = C/4 = 64.
// energy = Q Q^T (shared q/k weights -> symmetric); softmax over axis=-2
// == row softmax of E[m,:] -> flash attention with Q == K.
//
// Pipeline:
//   prep_x : x fp32 [B][C][N] -> xT bf16 [B][N][C]   (LDS tile transpose)
//   prep_w : weights fp32 -> bf16 (stacked [wqk;wv] and w_lbr), done once
//   proj   : pure-WMMA NT GEMM, fragments direct from L2-resident bf16 global
//            -> Q bf16 [B][N][64], V bf16 [B][C][N] (+bias)
//   flash  : online-softmax attention; V tiles staged to LDS by the Tensor
//            Data Mover (tensor_load_to_lds + s_wait_tensorcnt), TDM padding
//            produces the bank-conflict-free 72-elem LDS row stride.
//            -> xr = x - attn stored bf16 [B][N][C] (lbr's B layout)
//   lbr    : pure-WMMA NT GEMM + bias + BN(inference) + ReLU + residual.
// All matmuls: v_wmma_f32_16x16x32_bf16, fp32 accumulate.
// ---------------------------------------------------------------------------

#define CDIM 256
#define NDIM 4096
#define DK   64

typedef __attribute__((ext_vector_type(16))) __bf16 bf16x16;
typedef __attribute__((ext_vector_type(8)))  __bf16 bf16x8;
typedef __attribute__((ext_vector_type(8)))  float  f32x8;
typedef unsigned int u32x4 __attribute__((ext_vector_type(4)));
typedef int          i32x4 __attribute__((ext_vector_type(4)));
typedef int          i32x8 __attribute__((ext_vector_type(8)));

__device__ __forceinline__ f32x8 wmma_bf16(bf16x16 a, bf16x16 b, f32x8 c) {
  return __builtin_amdgcn_wmma_f32_16x16x32_bf16(
      /*neg_a=*/false, a, /*neg_b=*/false, b,
      /*c_mod=*/(short)0, c, /*reuse_a=*/false, /*reuse_b=*/false);
}

// A fragment (16x32, M x K), source laid out [row][K] K-contiguous.
__device__ __forceinline__ bf16x16 load_a(const __bf16* p0, int strideElems, int kk) {
  int lane = threadIdx.x & 31;
  const __bf16* p = p0 + (size_t)(lane & 15) * strideElems + kk + ((lane >> 4) << 3);
  bf16x8 lo = *(const bf16x8*)(p);
  bf16x8 hi = *(const bf16x8*)(p + 16);
  bf16x16 r;
#pragma unroll
  for (int i = 0; i < 8; ++i) { r[i] = lo[i]; r[i + 8] = hi[i]; }
  return r;
}

// B fragment (32x16, K x N), source laid out [col][K] K-contiguous.
__device__ __forceinline__ bf16x16 load_b(const __bf16* p0, int strideElems, int kk) {
  int lane = threadIdx.x & 31;
  const __bf16* p = p0 + (size_t)(lane & 15) * strideElems + kk + ((lane >> 4) << 3);
  bf16x8 lo = *(const bf16x8*)(p);
  bf16x8 hi = *(const bf16x8*)(p + 16);
  bf16x16 r;
#pragma unroll
  for (int i = 0; i < 8; ++i) { r[i] = lo[i]; r[i + 8] = hi[i]; }
  return r;
}

// ---------------------------------------------------------------------------
// TDM: 2-D tile load Global -> LDS, bf16 (data_size code 1 = 2B).
// D# packing per CDNA5 ISA.
// Pad: every 32 DWORDs (one 128B tile row) insert 4 DWORDs
// => LDS row stride = 64 + 8 = 72 bf16 elements.
// ---------------------------------------------------------------------------
__device__ __forceinline__ void tdm_load_2d_bf16(
    unsigned int lds_byte_off, unsigned long long gaddr,
    unsigned int tensor_d0, unsigned int tensor_d1,
    unsigned int tile_d0, unsigned int tile_d1, unsigned long long stride0) {
  u32x4 g0;
  g0[0] = 1u;                                              // count=1 (valid user D#)
  g0[1] = lds_byte_off;                                    // lds_addr
  g0[2] = (unsigned int)(gaddr & 0xffffffffu);             // global_addr[31:0]
  g0[3] = (unsigned int)((gaddr >> 32) & 0x01ffffffu)      // global_addr[56:32]
          | (2u << 30);                                    // type = 2 ("image")
  i32x8 g1;
  g1[0] = (int)((1u << 16)      // data_size = 1 -> 2 bytes
              | (1u << 20)      // pad_enable
              | (4u << 22)      // pad_interval: 2^(4+1)=32 DWORDs
              | (3u << 25));    // pad_amount: 3+1 = 4 DWORDs
  g1[1] = (int)((tensor_d0 & 0xffffu) << 16);                          // ab_addr=0 | td0 lo
  g1[2] = (int)((tensor_d0 >> 16) | ((tensor_d1 & 0xffffu) << 16));    // td0 hi | td1 lo
  g1[3] = (int)((tensor_d1 >> 16) | ((tile_d0 & 0xffffu) << 16));      // td1 hi | tile_d0
  g1[4] = (int)(tile_d1 & 0xffffu);                                    // tile_d1 | tile_d2=0
  g1[5] = (int)(unsigned int)(stride0 & 0xffffffffu);                  // stride0 lo
  g1[6] = (int)(unsigned int)((stride0 >> 32) & 0xffffu);              // stride0 hi | s1 lo=0
  g1[7] = 0;
  i32x4 z4 = {};
#if __clang_major__ >= 23
  i32x8 z8 = {};
  __builtin_amdgcn_tensor_load_to_lds(g0, g1, z4, z4, z8, 0);
#else
  __builtin_amdgcn_tensor_load_to_lds(g0, g1, z4, z4, 0);
#endif
}

// ---------------------------------------------------------------------------
// prep_x: x fp32 [B][C][N] -> xT bf16 [B][N][C], LDS tile transpose so both
// global read and write are coalesced. grid (N/64, C/64, B), block 256.
// ---------------------------------------------------------------------------
__global__ __launch_bounds__(256) void prep_x_kernel(const float* __restrict__ x,
                                                     __bf16* __restrict__ xT) {
  __shared__ __bf16 T[64 * 72];
  const int n0 = blockIdx.x * 64;
  const int c0 = blockIdx.y * 64;
  const int b  = blockIdx.z;
  const int t = threadIdx.x;
#pragma unroll
  for (int i = 0; i < 16; ++i) {
    int idx = t + i * 256;
    int n = idx & 63, c = idx >> 6;
    T[n * 72 + c] = (__bf16)x[((size_t)b * CDIM + c0 + c) * NDIM + n0 + n];
  }
  __syncthreads();
#pragma unroll
  for (int i = 0; i < 16; ++i) {
    int idx = t + i * 256;
    int c = idx & 63, n = idx >> 6;
    xT[((size_t)b * NDIM + n0 + n) * CDIM + c0 + c] = T[n * 72 + c];
  }
}

// ---------------------------------------------------------------------------
// prep_w: stacked [w_qk(64); w_v(256)] -> Wbf bf16 [320][256];
//         w_lbr -> WLbf bf16 [256][256].  grid 576 x 256.
// ---------------------------------------------------------------------------
__global__ __launch_bounds__(256) void prep_w_kernel(
    const float* __restrict__ w_qk, const float* __restrict__ w_v,
    const float* __restrict__ w_lbr, __bf16* __restrict__ Wbf,
    __bf16* __restrict__ WLbf) {
  int idx = blockIdx.x * 256 + threadIdx.x;
  if (idx < 320 * 256) {
    int r = idx >> 8, c = idx & 255;
    float v = (r < 64) ? w_qk[r * 256 + c] : w_v[(r - 64) * 256 + c];
    Wbf[idx] = (__bf16)v;
  } else {
    int i2 = idx - 320 * 256;
    if (i2 < 256 * 256) WLbf[i2] = (__bf16)w_lbr[i2];
  }
}

// ---------------------------------------------------------------------------
// proj: NT GEMM, fragments direct from global (L2/WGP$-resident bf16).
// Stacked weight rows [0,64) = Q (stored transposed), [64,320) = V (+bias).
// grid (N/64, 5, B), block 256 (8 waves), no LDS, no barriers.
// ---------------------------------------------------------------------------
__global__ __launch_bounds__(256) void proj_kernel(
    const __bf16* __restrict__ xT, const __bf16* __restrict__ Wbf,
    const float* __restrict__ b_v, __bf16* __restrict__ Qbf,
    __bf16* __restrict__ Vbf) {
  const int n0 = blockIdx.x * 64;
  const int m0 = blockIdx.y * 64;  // stacked-weight row base (0..256)
  const int b  = blockIdx.z;
  const int t = threadIdx.x, lane = t & 31, w = t >> 5;
  const int mt = w & 3, ct = w >> 2;  // wave: 16 rows x 32 cols
  f32x8 acc0 = {}, acc1 = {};
  const __bf16* Arow  = Wbf + (size_t)(m0 + mt * 16) * CDIM;
  const __bf16* Brow0 = xT + ((size_t)b * NDIM + n0 + ct * 32) * CDIM;
  const __bf16* Brow1 = Brow0 + (size_t)16 * CDIM;
#pragma unroll
  for (int kk = 0; kk < CDIM; kk += 32) {
    bf16x16 a  = load_a(Arow, CDIM, kk);
    bf16x16 b0 = load_b(Brow0, CDIM, kk);
    bf16x16 b1 = load_b(Brow1, CDIM, kk);
    acc0 = wmma_bf16(a, b0, acc0);
    acc1 = wmma_bf16(a, b1, acc1);
  }
  const int hi = lane >> 4;
  const int colb = n0 + ct * 32 + (lane & 15);
#pragma unroll
  for (int i = 0; i < 8; ++i) {
    int grow = m0 + mt * 16 + i + 8 * hi;
    float v0 = acc0[i], v1 = acc1[i];
    if (grow < 64) {  // Q rows: store transposed -> Q[b][n][q]
      Qbf[((size_t)b * NDIM + colb) * DK + grow]      = (__bf16)v0;
      Qbf[((size_t)b * NDIM + colb + 16) * DK + grow] = (__bf16)v1;
    } else {          // V rows: +bias -> V[b][c][n]
      int c = grow - 64;
      float bias = b_v[c];
      Vbf[((size_t)b * CDIM + c) * NDIM + colb]      = (__bf16)(v0 + bias);
      Vbf[((size_t)b * CDIM + c) * NDIM + colb + 16] = (__bf16)(v1 + bias);
    }
  }
}

// ---------------------------------------------------------------------------
// flash: one block = one 64-query tile; 8 waves; per-wave O accum 64m x 32c.
// V tile (256c x 64n bf16) DMA'd to LDS by TDM (wave 0), overlapping the
// S = QK^T WMMAs + both softmax passes; released by s_wait_tensorcnt+barrier.
// Writes xr = x - attn as bf16 in [B][N][C] (transposed: lbr's B layout).
// LDS: Ss 17408 + Ps 9216 + Vs 36864 + state 768 = 64256 B.
// ---------------------------------------------------------------------------
__global__ __launch_bounds__(256) void flash_kernel(
    const float* __restrict__ x, const __bf16* __restrict__ Qbf,
    const __bf16* __restrict__ Vbf, __bf16* __restrict__ XRt) {
  __shared__ float  Ss[64 * 68];
  __shared__ __bf16 Ps[64 * 72];
  __shared__ __bf16 Vs[256 * 72];   // TDM-padded rows: 64 data + 8 pad
  __shared__ float rowmax[64], rowsum[64], alphas[64];
  const int m0 = blockIdx.x * 64;
  const int b  = blockIdx.y;
  const int t = threadIdx.x, lane = t & 31, w = t >> 5;
  const int hi = lane >> 4;
  const int mt = w & 3, nh = w >> 2;  // S-phase patch
  const int c0 = w * 32;              // O columns owned by this wave
  if (t < 64) { rowmax[t] = -1e30f; rowsum[t] = 0.f; }
  const unsigned int ldsV = (unsigned int)(size_t)(void*)&Vs[0];

  // Q fragments for this wave's S row-tile, register-resident for all 64 iters
  const __bf16* Qbase = Qbf + ((size_t)b * NDIM + m0) * DK;
  bf16x16 qa0 = load_a(Qbase + mt * 16 * DK, DK, 0);
  bf16x16 qa1 = load_a(Qbase + mt * 16 * DK, DK, 32);

  f32x8 o[4][2];
  f32x8 z = {};
#pragma unroll
  for (int i = 0; i < 4; ++i) { o[i][0] = z; o[i][1] = z; }
  const float scale = 0.0625f;  // C^-0.5 = 1/16

  for (int kt = 0; kt < NDIM / 64; ++kt) {
    const int n0 = kt * 64;
    // ---- kick off async V-tile DMA (overlaps S + softmax below) ----
    if (w == 0) {
      const __bf16* vtile = Vbf + (size_t)b * CDIM * NDIM + n0;
      tdm_load_2d_bf16(ldsV, (unsigned long long)(size_t)vtile,
                       NDIM, CDIM, 64, CDIM, NDIM);
    }
    // ---- prefetch next K tile toward WGP ----
    if (kt + 1 < NDIM / 64) {
      const char* nk = (const char*)(Qbf + ((size_t)b * NDIM + n0 + 64) * DK);
      __builtin_prefetch(nk + t * 32, 0, 1);
    }
    // ---- S = (Q K^T) * scale for this wave's 16x32 patch (K == Q) ----
    const __bf16* Kbase = Qbf + ((size_t)b * NDIM + n0) * DK;
    f32x8 s0 = z, s1 = z;
#pragma unroll
    for (int kk = 0; kk < 64; kk += 32) {
      bf16x16 qa = (kk == 0) ? qa0 : qa1;
      bf16x16 kb0 = load_b(Kbase + (nh * 32) * DK, DK, kk);
      bf16x16 kb1 = load_b(Kbase + (nh * 32 + 16) * DK, DK, kk);
      s0 = wmma_bf16(qa, kb0, s0);
      s1 = wmma_bf16(qa, kb1, s1);
    }
#pragma unroll
    for (int i = 0; i < 8; ++i) {
      int r = mt * 16 + i + 8 * hi;
      int cc = nh * 32 + (lane & 15);
      Ss[r * 68 + cc]      = s0[i] * scale;
      Ss[r * 68 + cc + 16] = s1[i] * scale;
    }
    __syncthreads();
    // ---- online softmax: per-row new max + alpha rescale factor ----
    if (t < 64) {
      float mx = rowmax[t], nm = mx;
#pragma unroll 8
      for (int j = 0; j < 64; ++j) nm = fmaxf(nm, Ss[t * 68 + j]);
      float al = __expf(mx - nm);
      alphas[t] = al;
      rowmax[t] = nm;
      rowsum[t] *= al;
    }
    __syncthreads();
    // ---- exp -> P (bf16, K-contiguous rows) + partial row sums ----
    {
      int r = t >> 2;
      int cb = (t & 3) * 16;
      float nm = rowmax[r];
      float part = 0.f;
#pragma unroll
      for (int j = 0; j < 16; ++j) {
        float e = __expf(Ss[r * 68 + cb + j] - nm);
        Ps[r * 72 + cb + j] = (__bf16)e;
        part += e;
      }
      atomicAdd(&rowsum[r], part);
    }
    // ---- V tile must be resident before PV ----
    if (w == 0) __builtin_amdgcn_s_wait_tensorcnt(0);
    __syncthreads();
    // ---- rescale O by alpha, then O += P * V (V from LDS) ----
#pragma unroll
    for (int i2 = 0; i2 < 4; ++i2)
#pragma unroll
      for (int i = 0; i < 8; ++i) {
        float al = alphas[i2 * 16 + i + 8 * hi];
        o[i2][0][i] *= al;
        o[i2][1][i] *= al;
      }
#pragma unroll
    for (int kk = 0; kk < 64; kk += 32) {
      bf16x16 pa[4];
#pragma unroll
      for (int i2 = 0; i2 < 4; ++i2) pa[i2] = load_a(Ps + i2 * 16 * 72, 72, kk);
#pragma unroll
      for (int cs = 0; cs < 2; ++cs) {
        bf16x16 vb = load_b(Vs + (c0 + cs * 16) * 72, 72, kk);
#pragma unroll
        for (int i2 = 0; i2 < 4; ++i2) o[i2][cs] = wmma_bf16(pa[i2], vb, o[i2][cs]);
      }
    }
    __syncthreads();
  }
  // ---- finalize: xr[b][m][c] = x[b][c][m] - O/rowsum, bf16 transposed out ----
#pragma unroll
  for (int i2 = 0; i2 < 4; ++i2) {
#pragma unroll
    for (int cs = 0; cs < 2; ++cs) {
      int c = c0 + cs * 16 + (lane & 15);
#pragma unroll
      for (int i = 0; i < 8; ++i) {
        int rl = i2 * 16 + i + 8 * hi;
        float y = o[i2][cs][i] / rowsum[rl];
        float xv = x[((size_t)b * CDIM + c) * NDIM + m0 + rl];
        XRt[((size_t)b * NDIM + m0 + rl) * CDIM + c] = (__bf16)(xv - y);
      }
    }
  }
}

// ---------------------------------------------------------------------------
// lbr: h = w_lbr @ xr + b_lbr -> BN(inference) -> ReLU; out = x + h.
// Pure-WMMA NT GEMM, fragments direct from global. grid (N/64, 4, B).
// ---------------------------------------------------------------------------
__global__ __launch_bounds__(256) void lbr_kernel(
    const float* __restrict__ x, const __bf16* __restrict__ XRt,
    const __bf16* __restrict__ WLbf, const float* __restrict__ b_lbr,
    const float* __restrict__ gamma, const float* __restrict__ beta,
    const float* __restrict__ mean, const float* __restrict__ var,
    float* __restrict__ out) {
  const int n0 = blockIdx.x * 64;
  const int m0 = blockIdx.y * 64;
  const int b  = blockIdx.z;
  const int t = threadIdx.x, lane = t & 31, w = t >> 5;
  const int mt = w & 3, ct = w >> 2;
  f32x8 acc0 = {}, acc1 = {};
  const __bf16* Arow  = WLbf + (size_t)(m0 + mt * 16) * CDIM;
  const __bf16* Brow0 = XRt + ((size_t)b * NDIM + n0 + ct * 32) * CDIM;
  const __bf16* Brow1 = Brow0 + (size_t)16 * CDIM;
#pragma unroll
  for (int kk = 0; kk < CDIM; kk += 32) {
    bf16x16 a  = load_a(Arow, CDIM, kk);
    bf16x16 b0 = load_b(Brow0, CDIM, kk);
    bf16x16 b1 = load_b(Brow1, CDIM, kk);
    acc0 = wmma_bf16(a, b0, acc0);
    acc1 = wmma_bf16(a, b1, acc1);
  }
  const int hi = lane >> 4;
  const int colb = n0 + ct * 32 + (lane & 15);
#pragma unroll
  for (int i = 0; i < 8; ++i) {
    int row = m0 + mt * 16 + i + 8 * hi;
    float sc = gamma[row] * rsqrtf(var[row] + 1e-5f);
    float mn = mean[row], bt = beta[row], bl = b_lbr[row];
    float h0 = fmaxf((acc0[i] + bl - mn) * sc + bt, 0.f);
    float h1 = fmaxf((acc1[i] + bl - mn) * sc + bt, 0.f);
    size_t i0 = ((size_t)b * CDIM + row) * NDIM + colb;
    out[i0]      = x[i0] + h0;
    out[i0 + 16] = x[i0 + 16] + h1;
  }
}

// ---------------------------------------------------------------------------
extern "C" void kernel_launch(void* const* d_in, const int* in_sizes, int n_in,
                              void* d_out, int out_size, void* d_ws, size_t ws_size,
                              hipStream_t stream) {
  const float* x      = (const float*)d_in[0];
  const float* w_qk   = (const float*)d_in[1];
  const float* w_v    = (const float*)d_in[2];
  const float* b_v    = (const float*)d_in[3];
  const float* w_lbr  = (const float*)d_in[4];
  const float* b_lbr  = (const float*)d_in[5];
  const float* gammap = (const float*)d_in[6];
  const float* betap  = (const float*)d_in[7];
  const float* meanp  = (const float*)d_in[8];
  const float* varp   = (const float*)d_in[9];
  float* out = (float*)d_out;

  // workspace layout (bf16 buffers):
  //   Qbf 2MB | Vbf 8MB | XRt 8MB | xT 8MB | Wbf 160KB | WLbf 128KB
  char* ws = (char*)d_ws;
  __bf16* Qbf = (__bf16*)(ws);
  __bf16* Vbf = (__bf16*)(ws + ((size_t)2 << 20));
  __bf16* XRt = (__bf16*)(ws + ((size_t)10 << 20));
  __bf16* xT  = (__bf16*)(ws + ((size_t)18 << 20));
  __bf16* Wbf = (__bf16*)(ws + ((size_t)26 << 20));
  __bf16* WLbf = (__bf16*)(ws + ((size_t)26 << 20) + ((size_t)256 << 10));

  prep_x_kernel<<<dim3(NDIM / 64, CDIM / 64, 4), 256, 0, stream>>>(x, xT);
  prep_w_kernel<<<dim3(576), 256, 0, stream>>>(w_qk, w_v, w_lbr, Wbf, WLbf);
  proj_kernel<<<dim3(NDIM / 64, 5, 4), 256, 0, stream>>>(xT, Wbf, b_v, Qbf, Vbf);
  flash_kernel<<<dim3(NDIM / 64, 4), 256, 0, stream>>>(x, Qbf, Vbf, XRt);
  lbr_kernel<<<dim3(NDIM / 64, 4, 4), 256, 0, stream>>>(x, XRt, WLbf, b_lbr,
                                                        gammap, betap, meanp, varp, out);
}
